// EGNN_87643102642636
// MI455X (gfx1250) — matrix operand order repaired and checked by reading, not compiled, not run on hardware
//
#include <hip/hip_runtime.h>
#include <math.h>

#define NNODES 50000
#define NEDGES 800000
#define HD 128
#define NLAYERS 4
#define WBYTES (96 * 1024)        // edge/node staged weight block: 3 x 128x128 f16
#define EBYTES (32 * 1024)        // embed staged weight block: 1 x 128x128 f16
#define STAGE_BYTES (8 * 16 * HD * 2)

#if defined(__has_builtin)
#if __has_builtin(__builtin_amdgcn_tensor_load_to_lds)
#define EGNN_HAS_TDM 1
#endif
#endif
#ifndef EGNN_HAS_TDM
#define EGNN_HAS_TDM 0
#endif

typedef _Float16 half_t;
typedef __attribute__((ext_vector_type(16))) _Float16 v16h;
typedef __attribute__((ext_vector_type(8)))  _Float16 v8h;
typedef __attribute__((ext_vector_type(8)))  float    v8f;
typedef __attribute__((ext_vector_type(4)))  unsigned int u32x4;
typedef __attribute__((ext_vector_type(4)))  int i32x4;
typedef __attribute__((ext_vector_type(8)))  int i32x8;

static __device__ __forceinline__ float silu_f(float x) {
  return x / (1.0f + __expf(-x));
}
static __device__ __forceinline__ float sigmoid_f(float x) {
  return 1.0f / (1.0f + __expf(-x));
}

static __device__ __forceinline__ v8f wmma16(v16h a, v16h b, v8f c) {
  return __builtin_amdgcn_wmma_f32_16x16x32_f16(false, a, false, b, (short)0, c,
                                                false, false);
}

// A-fragment (16x32 f16, MxK): lanes 0-15 hold K 0..7 / 16..23; lanes 16-31
// hold K 8..15 / 24..31 within the 32-wide K tile kt.
static __device__ __forceinline__ v16h load_A_f16(const half_t* __restrict__ row,
                                                  int kt, int lane) {
  int base = kt * 32 + ((lane >> 4) << 3);
  v8h lo = *(const v8h*)(row + base);
  v8h hi = *(const v8h*)(row + base + 16);
  v16h a;
#pragma unroll
  for (int i = 0; i < 8; ++i) { a[i] = lo[i]; a[i + 8] = hi[i]; }
  return a;
}

static __device__ __forceinline__ v16h load_A_f32(const float* __restrict__ row,
                                                  int kt, int lane) {
  int base = kt * 32 + ((lane >> 4) << 3);
  v8f lo = *(const v8f*)(row + base);
  v8f hi = *(const v8f*)(row + base + 16);
  v16h a;
#pragma unroll
  for (int i = 0; i < 8; ++i) {
    a[i] = (_Float16)lo[i];
    a[i + 8] = (_Float16)hi[i];
  }
  return a;
}

// B fragments pre-packed: frag f = nt*4 + kt; lane's 16 halfs contiguous (32B).
static __device__ __forceinline__ v16h load_B(const half_t* __restrict__ packed,
                                              int frag, int lane) {
  return *(const v16h*)(packed + (((unsigned)frag * 32 + lane) << 4));
}

#if EGNN_HAS_TDM
// One TDM descriptor moves the whole weight block global -> LDS (1-D tile,
// 8-byte elements). Issued by one wave; TENSORcnt waited before the barrier.
static __device__ __forceinline__ void tdm_load(const void* g, char* lds_base,
                                                unsigned bytes) {
  unsigned long long ga = (unsigned long long)(uintptr_t)g;
  unsigned int laddr = (unsigned int)(uintptr_t)lds_base;  // LDS byte offset
  unsigned n8 = bytes >> 3;
  u32x4 g0;
  g0[0] = 1u;                                              // count=1 valid D#
  g0[1] = laddr;                                           // lds_addr
  g0[2] = (unsigned int)ga;                                // global_addr lo
  g0[3] = (unsigned int)((ga >> 32) & 0x01FFFFFFu) | (2u << 30);  // hi | type=2
  i32x8 g1;
  g1[0] = 3 << 16;                                         // data_size=8B
  g1[1] = (int)(n8 << 16);                                 // tensor_dim0 lo16
  g1[2] = 1 << 16;                                         // tensor_dim1=1
  g1[3] = (int)(n8 << 16);                                 // tile_dim0
  g1[4] = 0;
  g1[5] = (int)n8;                                         // tensor_dim0_stride
  g1[6] = 0;
  g1[7] = 0;
  i32x4 z4 = {0, 0, 0, 0};
#if defined(__clang_major__) && (__clang_major__ <= 22)
  __builtin_amdgcn_tensor_load_to_lds(g0, g1, z4, z4, 0);
#else
  i32x8 z8 = {0, 0, 0, 0, 0, 0, 0, 0};
  __builtin_amdgcn_tensor_load_to_lds(g0, g1, z4, z4, z8, 0);
#endif
  __builtin_amdgcn_s_wait_tensorcnt(0);
}
#endif

static __device__ __forceinline__ void stage_weights(const half_t* __restrict__ g,
                                                     char* lds_base, int tid,
                                                     unsigned bytes) {
#if EGNN_HAS_TDM
  if (tid < 32) tdm_load(g, lds_base, bytes);
#else
  const uint4* src = (const uint4*)g;
  uint4* dst = (uint4*)lds_base;
  for (unsigned i = tid; i < bytes / 16; i += 256) dst[i] = src[i];
#endif
  __syncthreads();
}

// ---------------- weight packer: 128x128 fp32 (KxN row-major) -> f16 B frags
__global__ void egnn_pack128(const float* __restrict__ W, half_t* __restrict__ out) {
  int tid = blockIdx.x * 256 + threadIdx.x;   // 16384 total
  int i = tid & 15;
  int lane = (tid >> 4) & 31;
  int f = tid >> 9;
  int kt = f & 3;
  int nt = f >> 2;
  int k = kt * 32 + ((lane >> 4) << 3) + (i & 7) + ((i >> 3) << 4);
  int n = (nt << 4) + (lane & 15);
  out[tid] = (half_t)W[k * HD + n];
}

__global__ void egnn_dist(const float* __restrict__ x, const int* __restrict__ st,
                          const int* __restrict__ en, float* __restrict__ d) {
  int i = blockIdx.x * 256 + threadIdx.x;
  if (i >= NEDGES) return;
  int a = st[i], b = en[i];
  float dx = x[a * 3 + 0] - x[b * 3 + 0];
  float dy = x[a * 3 + 1] - x[b * 3 + 1];
  float dz = x[a * 3 + 2] - x[b * 3 + 2];
  d[i] = sqrtf(dx * dx + dy * dy + dz * dz);
}

__global__ void egnn_zero(float* __restrict__ p, int n) {
  int i = blockIdx.x * 256 + threadIdx.x;
  if (i < n) p[i] = 0.0f;
}

// ---------------- node embedding; persistent, weights staged in LDS
__global__ __launch_bounds__(256) void egnn_embed(const float* __restrict__ hin,
                                                  const half_t* __restrict__ Bw,
                                                  const float* __restrict__ bb,
                                                  float* __restrict__ h,
                                                  half_t* __restrict__ hf,
                                                  int ntiles) {
  extern __shared__ char smem[];
  const int tid = threadIdx.x;
  const int lane = tid & 31;
  const int w = tid >> 5;
  stage_weights(Bw, smem, tid, EBYTES);
  const half_t* lw = (const half_t*)smem;
  const int r = lane & 15;
  const int hi8 = (lane >> 4) << 3;

  float bv[8];
#pragma unroll
  for (int nt = 0; nt < 8; ++nt) bv[nt] = bb[nt * 16 + r];
  const int nwaves = gridDim.x * 8;

  for (int tile = blockIdx.x * 8 + w; tile < ntiles; tile += nwaves) {
    const int n0 = tile * 16;
    v16h A[4];
#pragma unroll
    for (int kt = 0; kt < 4; ++kt)
      A[kt] = load_A_f32(hin + (size_t)(n0 + r) * HD, kt, lane);

    float* __restrict__ hp = h + (size_t)(n0 + hi8) * HD + r;
    half_t* __restrict__ hfp = hf + (size_t)(n0 + hi8) * HD + r;
#pragma unroll
    for (int nt = 0; nt < 8; ++nt) {
      v16h B[4];
#pragma unroll
      for (int kt = 0; kt < 4; ++kt) B[kt] = load_B(lw, nt * 4 + kt, lane);
      v8f c;
#pragma unroll
      for (int v = 0; v < 8; ++v) c[v] = bv[nt];
#pragma unroll
      for (int kt = 0; kt < 4; ++kt) c = wmma16(A[kt], B[kt], c);
#pragma unroll
      for (int v = 0; v < 8; ++v) {
        hp[v * HD + nt * 16] = c[v];
        hfp[v * HD + nt * 16] = (half_t)c[v];
      }
    }
  }
}

// ---------------- fused edge MLP + gate + scatter; persistent, weights in LDS
__global__ __launch_bounds__(256) void egnn_edge(
    const half_t* __restrict__ hf, const float* __restrict__ dvals,
    const int* __restrict__ st, const int* __restrict__ en,
    const half_t* __restrict__ Wpk,   // contiguous 96KB: [Bt | Bm | B2]
    const float* __restrict__ wlast, const float* __restrict__ b1,
    const float* __restrict__ b2, const float* __restrict__ winf,
    const float* __restrict__ binf, float* __restrict__ mi, int ntiles) {
  extern __shared__ char smem[];
  const int tid = threadIdx.x;
  const int lane = tid & 31;
  const int w = tid >> 5;
  stage_weights(Wpk, smem, tid, WBYTES);
  const half_t* lwT = (const half_t*)smem;
  const half_t* lwM = lwT + HD * HD;
  const half_t* lw2 = lwT + 2 * HD * HD;
  half_t* mt = (half_t*)(smem + WBYTES) + w * 16 * HD;
  const int r = lane & 15;
  const int hi8 = (lane >> 4) << 3;
  half_t* __restrict__ mtw = mt + hi8 * HD + r;   // C-layout store base

  // tile-invariant per-lane scalars
  float b1v[8], b2v[8], wlv[8], wiv[8];
#pragma unroll
  for (int nt = 0; nt < 8; ++nt) {
    b1v[nt] = b1[nt * 16 + r];
    b2v[nt] = b2[nt * 16 + r];
    wlv[nt] = wlast[nt * 16 + r];
    wiv[nt] = winf[nt * 16 + r];
  }
  const float bi = binf[0];
  const int nwaves = gridDim.x * 8;

  for (int tile = blockIdx.x * 8 + w; tile < ntiles; tile += nwaves) {
    const int e0 = tile * 16;
    const half_t* hs = hf + (size_t)st[e0 + r] * HD;
    const half_t* ht = hf + (size_t)en[e0 + r] * HD;

    v16h As[4], At[4];
#pragma unroll
    for (int kt = 0; kt < 4; ++kt) {
      As[kt] = load_A_f16(hs, kt, lane);
      At[kt] = load_A_f16(ht, kt, lane);
    }
    float dv[8];
    float* mp[8];   // per-row scatter base pointers (nt*16 folds to immediate)
#pragma unroll
    for (int v = 0; v < 8; ++v) {
      dv[v] = dvals[e0 + hi8 + v];
      mp[v] = mi + (size_t)st[e0 + hi8 + v] * HD + r;
    }

    // m = silu([h_i, h_j, d] @ fe1 + b1); d as rank-1 outer product
#pragma unroll
    for (int nt = 0; nt < 8; ++nt) {
      v16h bT[4], bM[4];
#pragma unroll
      for (int kt = 0; kt < 4; ++kt) {
        bT[kt] = load_B(lwT, nt * 4 + kt, lane);
        bM[kt] = load_B(lwM, nt * 4 + kt, lane);
      }
      v8f c;
#pragma unroll
      for (int v = 0; v < 8; ++v) c[v] = b1v[nt];
#pragma unroll
      for (int kt = 0; kt < 4; ++kt) {
        c = wmma16(As[kt], bT[kt], c);
        c = wmma16(At[kt], bM[kt], c);
      }
#pragma unroll
      for (int v = 0; v < 8; ++v) {
        float xv = c[v] + dv[v] * wlv[nt];
        mtw[v * HD + nt * 16] = (half_t)silu_f(xv);
      }
    }

    v16h A2[4];
#pragma unroll
    for (int kt = 0; kt < 4; ++kt) A2[kt] = load_A_f16(mt + r * HD, kt, lane);

    v8f m2[8];
    float part[8];
#pragma unroll
    for (int v = 0; v < 8; ++v) part[v] = 0.0f;
#pragma unroll
    for (int nt = 0; nt < 8; ++nt) {
      v16h B2f[4];
#pragma unroll
      for (int kt = 0; kt < 4; ++kt) B2f[kt] = load_B(lw2, nt * 4 + kt, lane);
      v8f c;
#pragma unroll
      for (int v = 0; v < 8; ++v) c[v] = b2v[nt];
#pragma unroll
      for (int kt = 0; kt < 4; ++kt) c = wmma16(A2[kt], B2f[kt], c);
#pragma unroll
      for (int v = 0; v < 8; ++v) {
        float xv = silu_f(c[v]);
        c[v] = xv;
        part[v] += xv * wiv[nt];
      }
      m2[nt] = c;
    }

    float eij[8];
#pragma unroll
    for (int v = 0; v < 8; ++v) {
      float p = part[v];
      p += __shfl_xor(p, 1, 32);
      p += __shfl_xor(p, 2, 32);
      p += __shfl_xor(p, 4, 32);
      p += __shfl_xor(p, 8, 32);
      eij[v] = sigmoid_f(p + bi);
    }

#pragma unroll
    for (int nt = 0; nt < 8; ++nt) {
#pragma unroll
      for (int v = 0; v < 8; ++v) {
        atomicAdd(mp[v] + nt * 16, eij[v] * m2[nt][v]);
      }
    }
  }
}

// ---------------- node update; persistent, weights in LDS
__global__ __launch_bounds__(256) void egnn_node(
    float* __restrict__ h, half_t* __restrict__ hf, const float* __restrict__ mi,
    const half_t* __restrict__ Wpk,   // contiguous 96KB: [B1t | B1b | B2]
    const float* __restrict__ b1, const float* __restrict__ b2, int ntiles) {
  extern __shared__ char smem[];
  const int tid = threadIdx.x;
  const int lane = tid & 31;
  const int w = tid >> 5;
  stage_weights(Wpk, smem, tid, WBYTES);
  const half_t* lw1t = (const half_t*)smem;
  const half_t* lw1b = lw1t + HD * HD;
  const half_t* lw2 = lw1t + 2 * HD * HD;
  half_t* ut = (half_t*)(smem + WBYTES) + w * 16 * HD;
  const int r = lane & 15;
  const int hi8 = (lane >> 4) << 3;
  half_t* __restrict__ utw = ut + hi8 * HD + r;

  float b1v[8], b2v[8];
#pragma unroll
  for (int nt = 0; nt < 8; ++nt) {
    b1v[nt] = b1[nt * 16 + r];
    b2v[nt] = b2[nt * 16 + r];
  }
  const int nwaves = gridDim.x * 8;

  for (int tile = blockIdx.x * 8 + w; tile < ntiles; tile += nwaves) {
    const int n0 = tile * 16;
    v16h Ah[4], Am[4];
#pragma unroll
    for (int kt = 0; kt < 4; ++kt) {
      Ah[kt] = load_A_f16(hf + (size_t)(n0 + r) * HD, kt, lane);
      Am[kt] = load_A_f32(mi + (size_t)(n0 + r) * HD, kt, lane);
    }

#pragma unroll
    for (int nt = 0; nt < 8; ++nt) {
      v16h bT[4], bB[4];
#pragma unroll
      for (int kt = 0; kt < 4; ++kt) {
        bT[kt] = load_B(lw1t, nt * 4 + kt, lane);
        bB[kt] = load_B(lw1b, nt * 4 + kt, lane);
      }
      v8f c;
#pragma unroll
      for (int v = 0; v < 8; ++v) c[v] = b1v[nt];
#pragma unroll
      for (int kt = 0; kt < 4; ++kt) {
        c = wmma16(Ah[kt], bT[kt], c);
        c = wmma16(Am[kt], bB[kt], c);
      }
#pragma unroll
      for (int v = 0; v < 8; ++v)
        utw[v * HD + nt * 16] = (half_t)silu_f(c[v]);
    }

    v16h A2[4];
#pragma unroll
    for (int kt = 0; kt < 4; ++kt) A2[kt] = load_A_f16(ut + r * HD, kt, lane);

    float* __restrict__ hp = h + (size_t)(n0 + hi8) * HD + r;
    half_t* __restrict__ hfp = hf + (size_t)(n0 + hi8) * HD + r;
#pragma unroll
    for (int nt = 0; nt < 8; ++nt) {
      v16h B2f[4];
#pragma unroll
      for (int kt = 0; kt < 4; ++kt) B2f[kt] = load_B(lw2, nt * 4 + kt, lane);
      v8f c;
#pragma unroll
      for (int v = 0; v < 8; ++v) c[v] = b2v[nt];
#pragma unroll
      for (int kt = 0; kt < 4; ++kt) c = wmma16(A2[kt], B2f[kt], c);
#pragma unroll
      for (int v = 0; v < 8; ++v) {
        float nh = hp[v * HD + nt * 16] + c[v];
        hp[v * HD + nt * 16] = nh;
        hfp[v * HD + nt * 16] = (half_t)nh;
      }
    }
  }
}

__global__ void egnn_copy_h(const float* __restrict__ h, float* __restrict__ out) {
  int i = blockIdx.x * 256 + threadIdx.x;
  if (i < NNODES * HD) out[i] = h[i];
}
__global__ void egnn_copy_e(const int* __restrict__ e, int* __restrict__ out) {
  int i = blockIdx.x * 256 + threadIdx.x;
  if (i < 2 * NEDGES) out[i] = e[i];
}

extern "C" void kernel_launch(void* const* d_in, const int* in_sizes, int n_in,
                              void* d_out, int out_size, void* d_ws, size_t ws_size,
                              hipStream_t stream) {
  (void)in_sizes; (void)n_in; (void)out_size; (void)ws_size;
  const float* h_in   = (const float*)d_in[0];
  const float* x      = (const float*)d_in[1];
  const int*   e      = (const int*)d_in[2];
  const float* emb_w  = (const float*)d_in[3];
  const float* emb_b  = (const float*)d_in[4];
  const float* fe1_w  = (const float*)d_in[5];
  const float* fe1_b  = (const float*)d_in[6];
  const float* fe2_w  = (const float*)d_in[7];
  const float* fe2_b  = (const float*)d_in[8];
  const float* finf_w = (const float*)d_in[9];
  const float* finf_b = (const float*)d_in[10];
  const float* fh1_w  = (const float*)d_in[11];
  const float* fh1_b  = (const float*)d_in[12];
  const float* fh2_w  = (const float*)d_in[13];
  const float* fh2_b  = (const float*)d_in[14];
  const int* st = e;
  const int* en = e + NEDGES;

  char* ws = (char*)d_ws;
  size_t off = 0;
  auto alloc = [&](size_t bytes) -> void* {
    void* p = ws + off;
    off = (off + bytes + 255) & ~(size_t)255;
    return p;
  };
  float*  hbuf = (float*)alloc((size_t)NNODES * HD * sizeof(float));
  half_t* hf16 = (half_t*)alloc((size_t)NNODES * HD * sizeof(half_t));
  float*  mi   = (float*)alloc((size_t)NNODES * HD * sizeof(float));
  float*  dbuf = (float*)alloc((size_t)NEDGES * sizeof(float));
  const size_t SEG = (size_t)HD * HD;            // 16384 halfs = 32KB
  half_t* pk_emb = (half_t*)alloc(SEG * sizeof(half_t));
  half_t *pk_edge[NLAYERS], *pk_node[NLAYERS];   // contiguous 96KB blocks
  for (int l = 0; l < NLAYERS; ++l) {
    pk_edge[l] = (half_t*)alloc(WBYTES);
    pk_node[l] = (half_t*)alloc(WBYTES);
  }

  // ---- pack weights to f16 WMMA B-fragment layout
  dim3 pk_grid(64), pk_blk(256);
  hipLaunchKernelGGL(egnn_pack128, pk_grid, pk_blk, 0, stream, emb_w, pk_emb);
  for (int l = 0; l < NLAYERS; ++l) {
    const float* fe1 = fe1_w + (size_t)l * 257 * HD;
    hipLaunchKernelGGL(egnn_pack128, pk_grid, pk_blk, 0, stream, fe1, pk_edge[l]);
    hipLaunchKernelGGL(egnn_pack128, pk_grid, pk_blk, 0, stream, fe1 + 128 * HD,
                       pk_edge[l] + SEG);
    hipLaunchKernelGGL(egnn_pack128, pk_grid, pk_blk, 0, stream,
                       fe2_w + (size_t)l * HD * HD, pk_edge[l] + 2 * SEG);
    const float* fh1 = fh1_w + (size_t)l * 256 * HD;
    hipLaunchKernelGGL(egnn_pack128, pk_grid, pk_blk, 0, stream, fh1, pk_node[l]);
    hipLaunchKernelGGL(egnn_pack128, pk_grid, pk_blk, 0, stream, fh1 + 128 * HD,
                       pk_node[l] + SEG);
    hipLaunchKernelGGL(egnn_pack128, pk_grid, pk_blk, 0, stream,
                       fh2_w + (size_t)l * HD * HD, pk_node[l] + 2 * SEG);
  }

  hipLaunchKernelGGL(egnn_dist, dim3((NEDGES + 255) / 256), dim3(256), 0, stream,
                     x, st, en, dbuf);
  const int node_tiles = NNODES / 16;
  hipLaunchKernelGGL(egnn_embed, dim3(256), dim3(256), EBYTES, stream,
                     h_in, pk_emb, emb_b, hbuf, hf16, node_tiles);

  const int edge_tiles = NEDGES / 16;
  const size_t shmem = WBYTES + STAGE_BYTES;     // 128 KB
  for (int l = 0; l < NLAYERS; ++l) {
    hipLaunchKernelGGL(egnn_zero, dim3((NNODES * HD + 255) / 256), dim3(256), 0,
                       stream, mi, NNODES * HD);
    const float* fe1 = fe1_w + (size_t)l * 257 * HD;
    hipLaunchKernelGGL(egnn_edge, dim3(1024), dim3(256), shmem, stream,
                       hf16, dbuf, st, en, pk_edge[l],
                       fe1 + 256 * HD, fe1_b + (size_t)l * HD,
                       fe2_b + (size_t)l * HD, finf_w + (size_t)l * HD,
                       finf_b + l, mi, edge_tiles);
    hipLaunchKernelGGL(egnn_node, dim3(256), dim3(256), shmem, stream,
                       hbuf, hf16, mi, pk_node[l],
                       fh1_b + (size_t)l * HD, fh2_b + (size_t)l * HD, node_tiles);
  }

  hipLaunchKernelGGL(egnn_copy_h, dim3((NNODES * HD + 255) / 256), dim3(256), 0,
                     stream, hbuf, (float*)d_out);
  hipLaunchKernelGGL(egnn_copy_e, dim3((2 * NEDGES + 255) / 256), dim3(256), 0,
                     stream, e, (int*)d_out + (size_t)NNODES * HD);
}
